// RIMCell_80041010528592
// MI455X (gfx1250) — matrix operand order repaired and checked by reading, not compiled, not run on hardware
//
#include <hip/hip_runtime.h>
#include <math.h>

// ---------------- problem constants ----------------
#define B_    2048
#define D_    1024
#define U_    1024
#define NR_   8
#define KTOP_ 5
#define NH_   4
#define KS_   64
#define VS_   1024
#define NCH_  4
#define CKS_  64
#define CVS_  256

typedef __attribute__((ext_vector_type(4)))  int    v4i;
typedef __attribute__((ext_vector_type(8)))  float  v8f;
typedef __attribute__((ext_vector_type(8)))  __bf16 v8bf;
typedef __attribute__((ext_vector_type(16))) __bf16 v16bf;

#define LP 40   // LDS row pitch in bf16 (32 data + 8 pad; 80B rows keep 16B alignment)

// gfx1250 async global->LDS path (ASYNCcnt), guarded per-toolchain
#if __has_builtin(__builtin_amdgcn_global_load_async_to_lds_b128) && \
    __has_builtin(__builtin_amdgcn_s_wait_asynccnt)
#define USE_ASYNC 1
#endif

// ---------------- helpers ----------------
// A-fragment K permutation within a 32-wide K block (v_wmma 16x16x32 bf16 A layout)
__device__ __forceinline__ int perm5(int k) {
    if (k < 8)  return k;
    if (k < 16) return k + 8;
    if (k < 24) return k - 8;
    return k;
}

// copy 16 bytes global -> LDS (async on gfx1250 when builtin available)
__device__ __forceinline__ void cp16(const __bf16* g, __bf16* l) {
#ifdef USE_ASYNC
    __builtin_amdgcn_global_load_async_to_lds_b128(
        (__attribute__((address_space(1))) v4i*)(g),
        (__attribute__((address_space(3))) v4i*)(l), 0, 0);
#else
    *(uint4*)l = *(const uint4*)g;
#endif
}
__device__ __forceinline__ void async_join() {
#ifdef USE_ASYNC
    __builtin_amdgcn_s_wait_asynccnt(0);
#endif
}

__device__ __forceinline__ v16bf ldfrag(const __bf16* p) {
    v8bf lo = *(const v8bf*)p;
    v8bf hi = *(const v8bf*)(p + 8);
    v16bf r;
#pragma unroll
    for (int i = 0; i < 8; ++i) { r[i] = lo[i]; r[8 + i] = hi[i]; }
    return r;
}

__device__ __forceinline__ v8f wmma_bf16(v16bf a, v16bf b, v8f c) {
    return __builtin_amdgcn_wmma_f32_16x16x32_bf16(false, a, false, b, (short)0, c, false, false);
}

enum { EPI_NONE = 0, EPI_BIAS = 1, EPI_FINAL = 2 };

// ---------------- generic bf16 WMMA GEMM ----------------
// C[z][row,col] = sum_k A[row, z*azs + k] * Bt[z][col][k]   (Bt pre-transposed [n][k])
// A is bf16 with perm5 applied per 32-block of K; Bt is bf16 natural-K.
// Block tile 128x128, 256 threads = 8 waves (4M x 2N); wave tile 32x64 (2x4 frags).
template <int EPI, int WB16>
__global__ __launch_bounds__(256)
void gemm_bf16(const __bf16* __restrict__ A, long ars, long azs,
               const __bf16* __restrict__ Bt, long bzs,
               float* __restrict__ C, long crs, long czs,
               __bf16* __restrict__ Cb,          // optional bf16 perm copy (same strides)
               int Kd,
               const float* __restrict__ bias,
               const float* __restrict__ hnew,
               const float* __restrict__ hsp,
               const float* __restrict__ maskf)
{
    __shared__ __bf16 lsA[128 * LP];
    __shared__ __bf16 lsB[128 * LP];

    const int t   = threadIdx.x;
    const int z   = blockIdx.z;
    const int bm0 = blockIdx.y * 128;
    const int bn0 = blockIdx.x * 128;

    const __bf16* Az = A + (long)z * azs;
    const __bf16* Bz = Bt + (long)z * bzs;

    const int wave = t >> 5, lane = t & 31;
    const int wm = wave & 3, wn = wave >> 2;
    const int lm = lane & 15, lg = lane >> 4;

    v8f acc[2][4];
#pragma unroll
    for (int f = 0; f < 2; ++f)
#pragma unroll
        for (int j = 0; j < 4; ++j)
#pragma unroll
            for (int e = 0; e < 8; ++e) acc[f][j][e] = 0.f;

    const int sr = t >> 1;          // 0..127: A row / B n-row
    const int sc = (t & 1) * 16;    // element offset in 32-wide K slice

    for (int k0 = 0; k0 < Kd; k0 += 32) {
        __syncthreads();
        {
            const __bf16* ag = Az + (long)(bm0 + sr) * ars + k0 + sc;
            const __bf16* bg = Bz + (long)(bn0 + sr) * Kd + k0 + sc;
#ifndef USE_ASYNC
            if (k0 + 32 < Kd) { __builtin_prefetch(ag + 32, 0, 0); __builtin_prefetch(bg + 32, 0, 0); }
#endif
            cp16(ag,     &lsA[sr * LP + sc]);
            cp16(ag + 8, &lsA[sr * LP + sc + 8]);
            cp16(bg,     &lsB[sr * LP + (t & 1) * 8]);
            cp16(bg + 8, &lsB[sr * LP + 16 + (t & 1) * 8]);
        }
        async_join();
        __syncthreads();

        v16bf a0 = ldfrag(&lsA[(wm * 32      + lm) * LP + lg * 16]);
        v16bf a1 = ldfrag(&lsA[(wm * 32 + 16 + lm) * LP + lg * 16]);
        v16bf b0 = ldfrag(&lsB[(wn * 64      + lm) * LP + lg * 16]);
        v16bf b1 = ldfrag(&lsB[(wn * 64 + 16 + lm) * LP + lg * 16]);
        v16bf b2 = ldfrag(&lsB[(wn * 64 + 32 + lm) * LP + lg * 16]);
        v16bf b3 = ldfrag(&lsB[(wn * 64 + 48 + lm) * LP + lg * 16]);

        acc[0][0] = wmma_bf16(a0, b0, acc[0][0]);
        acc[0][1] = wmma_bf16(a0, b1, acc[0][1]);
        acc[0][2] = wmma_bf16(a0, b2, acc[0][2]);
        acc[0][3] = wmma_bf16(a0, b3, acc[0][3]);
        acc[1][0] = wmma_bf16(a1, b0, acc[1][0]);
        acc[1][1] = wmma_bf16(a1, b1, acc[1][1]);
        acc[1][2] = wmma_bf16(a1, b2, acc[1][2]);
        acc[1][3] = wmma_bf16(a1, b3, acc[1][3]);
    }

#pragma unroll
    for (int f = 0; f < 2; ++f) {
#pragma unroll
        for (int j = 0; j < 4; ++j) {
            const int col  = bn0 + wn * 64 + j * 16 + lm;
            const int pcol = (col & ~31) | perm5(col & 31);
            float bvv = 0.f;
            if (EPI == EPI_BIAS) bvv = bias[col];
#pragma unroll
            for (int e = 0; e < 8; ++e) {
                const int row = bm0 + wm * 32 + f * 16 + 8 * lg + e;
                float v = acc[f][j][e];
                if (EPI == EPI_BIAS) v += bvv;
                const long cbase = (long)row * crs + (long)z * czs;
                if (EPI == EPI_FINAL) {
                    const float m = maskf[row * NR_ + z];
                    v = (v + hnew[cbase + col]) * m + hsp[cbase + col] * (1.f - m);
                }
                if (C) C[cbase + col] = v;
                if (WB16) Cb[cbase + pcol] = (__bf16)v;
            }
        }
    }
}

// ---------------- fused GroupAlphatCell ----------------
// pre_r/i = c0[b,n]*(v2b @ Wi2h[n])_{col,col+U} + (hs_n @ Wh2h[n])_{col,col+U} + c1*bias_i
// h_new = tanh(pre_i)*sigmoid(pre_r) + hs*(1-sigmoid(pre_r))
// Tile 128M x 32N; separate accumulators for the two GEMM phases, scale in epilogue.
__global__ __launch_bounds__(256)
void cell_wmma(const __bf16* __restrict__ v2bb, const __bf16* __restrict__ hsb,
               const __bf16* __restrict__ Wi2hT, const __bf16* __restrict__ Wh2hT,
               const float* __restrict__ c0v, const float* __restrict__ c1v,
               const float* __restrict__ biasi, const float* __restrict__ hsf,
               float* __restrict__ hnew, __bf16* __restrict__ hnewb)
{
    __shared__ __bf16 lsA[128 * LP];
    __shared__ __bf16 lsBr[32 * LP];
    __shared__ __bf16 lsBi[32 * LP];

    const int t = threadIdx.x;
    const int n = blockIdx.z;
    const int bm0 = blockIdx.y * 128;
    const int bn0 = blockIdx.x * 32;

    const int wave = t >> 5, lane = t & 31;
    const int wm = wave & 3, wn = wave >> 2;
    const int lm = lane & 15, lg = lane >> 4;

    v8f vr[2], vi[2], hr[2], hi[2];
#pragma unroll
    for (int f = 0; f < 2; ++f)
#pragma unroll
        for (int e = 0; e < 8; ++e) { vr[f][e] = 0.f; vi[f][e] = 0.f; hr[f][e] = 0.f; hi[f][e] = 0.f; }

    const int sr = t >> 1;
    const int sc = (t & 1) * 16;
    const int brow = t >> 2;                 // 0..63 (32 r-rows then 32 i-rows)
    const int bc = (t & 3) * 8;
    const long bo = (brow < 32) ? (long)(bn0 + brow) : (long)(U_ + bn0 + (brow - 32));
    __bf16* lb = (brow < 32) ? &lsBr[brow * LP] : &lsBi[(brow - 32) * LP];
    const long wstride = (long)n * 2 * U_ * U_;   // [n][2048][1024]

    // ---- phase 1: v2b @ Wi2h[n] ----
    for (int k0 = 0; k0 < U_; k0 += 32) {
        __syncthreads();
        {
            const __bf16* ag = v2bb + (long)(bm0 + sr) * VS_ + k0 + sc;
            cp16(ag,     &lsA[sr * LP + sc]);
            cp16(ag + 8, &lsA[sr * LP + sc + 8]);
            const __bf16* wg = Wi2hT + wstride + bo * U_ + k0 + bc;
            cp16(wg, lb + perm5(bc));
        }
        async_join();
        __syncthreads();
        v16bf a0 = ldfrag(&lsA[(wm * 32      + lm) * LP + lg * 16]);
        v16bf a1 = ldfrag(&lsA[(wm * 32 + 16 + lm) * LP + lg * 16]);
        v16bf br = ldfrag(&lsBr[(wn * 16 + lm) * LP + lg * 16]);
        v16bf bi = ldfrag(&lsBi[(wn * 16 + lm) * LP + lg * 16]);
        vr[0] = wmma_bf16(a0, br, vr[0]);
        vr[1] = wmma_bf16(a1, br, vr[1]);
        vi[0] = wmma_bf16(a0, bi, vi[0]);
        vi[1] = wmma_bf16(a1, bi, vi[1]);
    }
    // ---- phase 2: hs[:,n,:] @ Wh2h[n] ----
    for (int k0 = 0; k0 < U_; k0 += 32) {
        __syncthreads();
        {
            const __bf16* ag = hsb + (long)(bm0 + sr) * (NR_ * U_) + (long)n * U_ + k0 + sc;
            cp16(ag,     &lsA[sr * LP + sc]);
            cp16(ag + 8, &lsA[sr * LP + sc + 8]);
            const __bf16* wg = Wh2hT + wstride + bo * U_ + k0 + bc;
            cp16(wg, lb + perm5(bc));
        }
        async_join();
        __syncthreads();
        v16bf a0 = ldfrag(&lsA[(wm * 32      + lm) * LP + lg * 16]);
        v16bf a1 = ldfrag(&lsA[(wm * 32 + 16 + lm) * LP + lg * 16]);
        v16bf br = ldfrag(&lsBr[(wn * 16 + lm) * LP + lg * 16]);
        v16bf bi = ldfrag(&lsBi[(wn * 16 + lm) * LP + lg * 16]);
        hr[0] = wmma_bf16(a0, br, hr[0]);
        hr[1] = wmma_bf16(a1, br, hr[1]);
        hi[0] = wmma_bf16(a0, bi, hi[0]);
        hi[1] = wmma_bf16(a1, bi, hi[1]);
    }

    const int col  = bn0 + wn * 16 + lm;
    const int pcol = (col & ~31) | perm5(col & 31);
    const float brb = biasi[n * (2 * U_) + col];
    const float bib = biasi[n * (2 * U_) + U_ + col];
#pragma unroll
    for (int f = 0; f < 2; ++f) {
#pragma unroll
        for (int e = 0; e < 8; ++e) {
            const int row = bm0 + wm * 32 + f * 16 + 8 * lg + e;
            const float cc0 = c0v[row * NR_ + n];
            const float cc1 = c1v[row * NR_ + n];
            const float pr = cc0 * vr[f][e] + hr[f][e] + cc1 * brb;
            const float pi = cc0 * vi[f][e] + hi[f][e] + cc1 * bib;
            const float alpha = 1.f / (1.f + __expf(-pr));
            const float hh = tanhf(pi);
            const long idx = (long)row * (NR_ * U_) + (long)n * U_;
            const float hn = hh * alpha + hsf[idx + col] * (1.f - alpha);
            hnew[idx + col] = hn;
            hnewb[idx + pcol] = (__bf16)hn;
        }
    }
}

// ---------------- preprocessing kernels ----------------
// f32 [Z][K][N] -> bf16 [Z][N][K] (weights, column-major for B feed)
__global__ void transpose_cvt(const float* __restrict__ in, __bf16* __restrict__ out, int K, int N)
{
    __shared__ float tile[32][33];
    const int z = blockIdx.z;
    const float* inz = in + (long)z * K * N;
    __bf16* outz = out + (long)z * N * K;
    const int n0 = blockIdx.x * 32, k0 = blockIdx.y * 32;
    const int tx = threadIdx.x & 31, ty = threadIdx.x >> 5;
#pragma unroll
    for (int i = 0; i < 4; ++i)
        tile[ty + 8 * i][tx] = inz[(long)(k0 + ty + 8 * i) * N + n0 + tx];
    __syncthreads();
#pragma unroll
    for (int i = 0; i < 4; ++i)
        outz[(long)(n0 + ty + 8 * i) * K + k0 + tx] = (__bf16)tile[tx][ty + 8 * i];
}

// head-mean of Wv, transposed+converted: outT[v][d] = bf16(mean_h Wv[d][h*VS+v])
__global__ void wvmean_t(const float* __restrict__ Wv, __bf16* __restrict__ outT)
{
    __shared__ float tile[32][33];
    const int v0 = blockIdx.x * 32, d0 = blockIdx.y * 32;
    const int tx = threadIdx.x & 31, ty = threadIdx.x >> 5;
#pragma unroll
    for (int i = 0; i < 4; ++i) {
        const long d = d0 + ty + 8 * i;
        const float* p = Wv + d * (NH_ * VS_) + v0 + tx;
        tile[ty + 8 * i][tx] = 0.25f * (p[0] + p[VS_] + p[2 * VS_] + p[3 * VS_]);
    }
    __syncthreads();
#pragma unroll
    for (int i = 0; i < 4; ++i)
        outT[(long)(v0 + ty + 8 * i) * D_ + d0 + tx] = (__bf16)tile[tx][ty + 8 * i];
}

__global__ void bvmean_k(const float* __restrict__ bv, float* __restrict__ bvmean)
{
    const int i = blockIdx.x * 256 + threadIdx.x;
    if (i < VS_) bvmean[i] = 0.25f * (bv[i] + bv[i + VS_] + bv[i + 2 * VS_] + bv[i + 3 * VS_]);
}

// f32 -> bf16 with perm5 applied within each 32-aligned block (A-side activations)
__global__ void cvt_perm(const float* __restrict__ in, __bf16* __restrict__ out, long nElem)
{
    const long i = (long)blockIdx.x * 256 + threadIdx.x;
    if (i < nElem) out[(i & ~31L) | perm5((int)(i & 31))] = (__bf16)in[i];
}

__global__ void biasi_kernel(const float* __restrict__ bvmean, const float* __restrict__ Wi2h,
                             float* __restrict__ biasi)
{
    const int idx = blockIdx.x * 256 + threadIdx.x;
    if (idx >= NR_ * 2 * U_) return;
    const int n = idx >> 11, o = idx & 2047;
    const float* w = Wi2h + (long)n * VS_ * 2 * U_ + o;
    float acc = 0.f;
    for (int v = 0; v < VS_; ++v) acc += bvmean[v] * w[(long)v * 2048];
    biasi[idx] = acc;
}

// ---------------- attention scalar path ----------------
__global__ void score_kernel(const float* __restrict__ q, const float* __restrict__ kx,
                             const float* __restrict__ bk,
                             float* __restrict__ s0, float* __restrict__ s1)
{
    const int p = blockIdx.x * 8 + (threadIdx.x >> 5);
    const int lane = threadIdx.x & 31;
    const int b = p >> 3, n = p & 7;
    float a0 = 0.f, a1 = 0.f;
    for (int i = lane; i < NH_ * KS_; i += 32) {
        const float qq = q[(long)b * (NR_ * NH_ * KS_) + n * (NH_ * KS_) + i];
        a0 += qq * kx[(long)b * (NH_ * KS_) + i];
        a1 += qq * bk[i];
    }
    for (int off = 16; off > 0; off >>= 1) {
        a0 += __shfl_xor(a0, off, 32);
        a1 += __shfl_xor(a1, off, 32);
    }
    if (lane == 0) {
        const float sc = 1.f / (NH_ * 8.f);
        s0[p] = a0 * sc;
        s1[p] = a1 * sc;
    }
}

__global__ void topk_kernel(const float* __restrict__ s0, const float* __restrict__ s1,
                            float* __restrict__ maskf, float* __restrict__ c0, float* __restrict__ c1)
{
    const int b = blockIdx.x * blockDim.x + threadIdx.x;
    if (b >= B_) return;
    float sv[NR_]; bool sel[NR_];
#pragma unroll
    for (int n = 0; n < NR_; ++n) { sv[n] = s0[b * NR_ + n]; sel[n] = false; }
#pragma unroll
    for (int it = 0; it < KTOP_; ++it) {
        int bi = 0; float best = -3.4e38f;
#pragma unroll
        for (int n = 0; n < NR_; ++n)
            if (!sel[n] && sv[n] > best) { best = sv[n]; bi = n; }
        sel[bi] = true;
    }
#pragma unroll
    for (int n = 0; n < NR_; ++n) {
        const float m = sel[n] ? 1.f : 0.f;
        const float a = sv[n], bb = s1[b * NR_ + n];
        const float mx = fmaxf(a, bb);
        const float e0 = __expf(a - mx), e1 = __expf(bb - mx);
        const float p0 = e0 / (e0 + e1);
        maskf[b * NR_ + n] = m;
        c0[b * NR_ + n] = p0 * m;
        c1[b * NR_ + n] = (1.f - p0) * m;
    }
}

// communication attention per batch element -> ctx (bf16, perm5 on K for next GEMM)
__global__ __launch_bounds__(256)
void comm_attn(const float* __restrict__ ck, const float* __restrict__ cv,
               const float* __restrict__ cq, const float* __restrict__ maskf,
               __bf16* __restrict__ ctxb)
{
    __shared__ float sck[NR_ * NCH_ * CKS_];
    __shared__ float scq[NR_ * NCH_ * CKS_];
    __shared__ float scv[NR_ * NCH_ * CVS_];
    __shared__ float scp[NCH_ * NR_ * NR_];
    const int b = blockIdx.x;
    const int t = threadIdx.x;

    for (int i = t; i < NR_ * NCH_ * CKS_; i += 256) {
        sck[i] = ck[(long)b * (NR_ * NCH_ * CKS_) + i];
        scq[i] = cq[(long)b * (NR_ * NCH_ * CKS_) + i];
    }
    for (int i = t; i < NR_ * NCH_ * CVS_; i += 256)
        scv[i] = cv[(long)b * (NR_ * NCH_ * CVS_) + i];
    __syncthreads();
    {
        const int h = t >> 6, n = (t >> 3) & 7, m = t & 7;
        float acc = 0.f;
        for (int k = 0; k < CKS_; ++k)
            acc += scq[n * 256 + h * CKS_ + k] * sck[m * 256 + h * CKS_ + k];
        scp[t] = acc * 0.125f;
    }
    __syncthreads();
    if (t < NCH_ * NR_) {
        const int h = t >> 3, n = t & 7;
        float mx = -3.4e38f;
#pragma unroll
        for (int m = 0; m < NR_; ++m) mx = fmaxf(mx, scp[h * 64 + n * 8 + m]);
        float sum = 0.f; float e[NR_];
#pragma unroll
        for (int m = 0; m < NR_; ++m) { e[m] = __expf(scp[h * 64 + n * 8 + m] - mx); sum += e[m]; }
        const float msk = maskf[b * NR_ + n] / sum;
#pragma unroll
        for (int m = 0; m < NR_; ++m) scp[h * 64 + n * 8 + m] = e[m] * msk;
    }
    __syncthreads();
    for (int idx = t; idx < NR_ * NCH_ * CVS_; idx += 256) {
        const int n = idx >> 10, r = idx & 1023, h = r >> 8, vv = r & 255;
        float acc = 0.f;
#pragma unroll
        for (int m = 0; m < NR_; ++m)
            acc += scp[h * 64 + n * 8 + m] * scv[m * 1024 + h * 256 + vv];
        ctxb[(long)b * (NR_ * NCH_ * CVS_) + ((idx & ~31) | perm5(idx & 31))] = (__bf16)acc;
    }
}

// ---------------- launcher ----------------
extern "C" void kernel_launch(void* const* d_in, const int* in_sizes, int n_in,
                              void* d_out, int out_size, void* d_ws, size_t ws_size,
                              hipStream_t stream)
{
    const float* x    = (const float*)d_in[0];
    const float* hs   = (const float*)d_in[1];
    const float* Wk   = (const float*)d_in[2];
    const float* bk   = (const float*)d_in[3];
    const float* Wv   = (const float*)d_in[4];
    const float* bv   = (const float*)d_in[5];
    const float* Wq   = (const float*)d_in[6];
    const float* Wi2h = (const float*)d_in[7];
    const float* Wh2h = (const float*)d_in[8];
    const float* Wck  = (const float*)d_in[9];
    const float* Wcv  = (const float*)d_in[10];
    const float* Wcq  = (const float*)d_in[11];
    const float* Wco  = (const float*)d_in[12];
    float* out = (float*)d_out;

    char* base = (char*)d_ws;
    size_t off = 0;
    auto allocf = [&](size_t n) { float* p = (float*)(base + off); off += ((n * 4 + 255) & ~(size_t)255); return p; };
    auto allocb = [&](size_t n) { __bf16* p = (__bf16*)(base + off); off += ((n * 2 + 255) & ~(size_t)255); return p; };

    // f32 buffers
    float* kx     = allocf((size_t)B_ * NH_ * KS_);
    float* qb     = allocf((size_t)B_ * NR_ * NH_ * KS_);
    float* s0b    = allocf(B_ * NR_);
    float* s1b    = allocf(B_ * NR_);
    float* maskf  = allocf(B_ * NR_);
    float* c0b    = allocf(B_ * NR_);
    float* c1b    = allocf(B_ * NR_);
    float* bvmean = allocf(VS_);
    float* biasi  = allocf(NR_ * 2 * U_);
    float* hnew   = allocf((size_t)B_ * NR_ * U_);
    float* ckb    = allocf((size_t)B_ * NR_ * NCH_ * CKS_);
    float* cqb    = allocf((size_t)B_ * NR_ * NCH_ * CKS_);
    float* cvb    = allocf((size_t)B_ * NR_ * NCH_ * CVS_);
    // bf16 buffers (A-side perm5 layout / B-side [n][k] transposed)
    __bf16* xb      = allocb((size_t)B_ * D_);
    __bf16* hsb     = allocb((size_t)B_ * NR_ * U_);
    __bf16* v2bb    = allocb((size_t)B_ * VS_);
    __bf16* hnewb   = allocb((size_t)B_ * NR_ * U_);
    __bf16* ctxb    = allocb((size_t)B_ * NR_ * NCH_ * CVS_);
    __bf16* WkT     = allocb((size_t)(NH_ * KS_) * D_);
    __bf16* wvmT    = allocb((size_t)VS_ * D_);
    __bf16* WqT     = allocb((size_t)NR_ * (NH_ * KS_) * U_);
    __bf16* Wi2hT   = allocb((size_t)NR_ * 2 * U_ * VS_);
    __bf16* Wh2hT   = allocb((size_t)NR_ * 2 * U_ * U_);
    __bf16* WckT    = allocb((size_t)NR_ * (NCH_ * CKS_) * U_);
    __bf16* WcqT    = allocb((size_t)NR_ * (NCH_ * CKS_) * U_);
    __bf16* WcvT    = allocb((size_t)NR_ * (NCH_ * CVS_) * U_);
    __bf16* WcoT    = allocb((size_t)NR_ * U_ * (NCH_ * CVS_));

    // ---- weight / activation preprocessing (once per call) ----
    transpose_cvt<<<dim3((NH_ * KS_) / 32, D_ / 32, 1), 256, 0, stream>>>(Wk, WkT, D_, NH_ * KS_);
    wvmean_t<<<dim3(VS_ / 32, D_ / 32, 1), 256, 0, stream>>>(Wv, wvmT);
    bvmean_k<<<(VS_ + 255) / 256, 256, 0, stream>>>(bv, bvmean);
    transpose_cvt<<<dim3((NH_ * KS_) / 32, U_ / 32, NR_), 256, 0, stream>>>(Wq, WqT, U_, NH_ * KS_);
    transpose_cvt<<<dim3((2 * U_) / 32, VS_ / 32, NR_), 256, 0, stream>>>(Wi2h, Wi2hT, VS_, 2 * U_);
    transpose_cvt<<<dim3((2 * U_) / 32, U_ / 32, NR_), 256, 0, stream>>>(Wh2h, Wh2hT, U_, 2 * U_);
    transpose_cvt<<<dim3((NCH_ * CKS_) / 32, U_ / 32, NR_), 256, 0, stream>>>(Wck, WckT, U_, NCH_ * CKS_);
    transpose_cvt<<<dim3((NCH_ * CKS_) / 32, U_ / 32, NR_), 256, 0, stream>>>(Wcq, WcqT, U_, NCH_ * CKS_);
    transpose_cvt<<<dim3((NCH_ * CVS_) / 32, U_ / 32, NR_), 256, 0, stream>>>(Wcv, WcvT, U_, NCH_ * CVS_);
    transpose_cvt<<<dim3(U_ / 32, (NCH_ * CVS_) / 32, NR_), 256, 0, stream>>>(Wco, WcoT, NCH_ * CVS_, U_);
    cvt_perm<<<((size_t)B_ * D_) / 256, 256, 0, stream>>>(x, xb, (long)B_ * D_);
    cvt_perm<<<((size_t)B_ * NR_ * U_) / 256, 256, 0, stream>>>(hs, hsb, (long)B_ * NR_ * U_);
    biasi_kernel<<<(NR_ * 2 * U_ + 255) / 256, 256, 0, stream>>>(bvmean, Wi2h, biasi);

    // 1) kx = x @ Wk + bk
    gemm_bf16<EPI_BIAS, 0><<<dim3((NH_ * KS_) / 128, B_ / 128, 1), 256, 0, stream>>>(
        xb, D_, 0, WkT, 0, kx, NH_ * KS_, 0, nullptr, D_, bk, nullptr, nullptr, nullptr);
    // 2) v2b = x @ Wv_mean + bv_mean  (bf16 perm output only)
    gemm_bf16<EPI_BIAS, 1><<<dim3(VS_ / 128, B_ / 128, 1), 256, 0, stream>>>(
        xb, D_, 0, wvmT, 0, nullptr, VS_, 0, v2bb, D_, bvmean, nullptr, nullptr, nullptr);
    // 3) q[:,n,:] = hs[:,n,:] @ Wq[n]
    gemm_bf16<EPI_NONE, 0><<<dim3((NH_ * KS_) / 128, B_ / 128, NR_), 256, 0, stream>>>(
        hsb, NR_ * U_, U_, WqT, (long)(NH_ * KS_) * U_, qb, NR_ * NH_ * KS_, NH_ * KS_,
        nullptr, U_, nullptr, nullptr, nullptr, nullptr);
    // 4) logits, top-k mask, softmax coefficients
    score_kernel<<<(B_ * NR_) / 8, 256, 0, stream>>>(qb, kx, bk, s0b, s1b);
    topk_kernel<<<(B_ + 255) / 256, 256, 0, stream>>>(s0b, s1b, maskf, c0b, c1b);
    // 5) fused gated cell -> h_new (f32 + bf16 perm)
    cell_wmma<<<dim3(U_ / 32, B_ / 128, NR_), 256, 0, stream>>>(
        v2bb, hsb, Wi2hT, Wh2hT, c0b, c1b, biasi, hs, hnew, hnewb);
    // 6) comm projections
    gemm_bf16<EPI_NONE, 0><<<dim3((NCH_ * CKS_) / 128, B_ / 128, NR_), 256, 0, stream>>>(
        hnewb, NR_ * U_, U_, WckT, (long)(NCH_ * CKS_) * U_, ckb, NR_ * NCH_ * CKS_, NCH_ * CKS_,
        nullptr, U_, nullptr, nullptr, nullptr, nullptr);
    gemm_bf16<EPI_NONE, 0><<<dim3((NCH_ * CKS_) / 128, B_ / 128, NR_), 256, 0, stream>>>(
        hnewb, NR_ * U_, U_, WcqT, (long)(NCH_ * CKS_) * U_, cqb, NR_ * NCH_ * CKS_, NCH_ * CKS_,
        nullptr, U_, nullptr, nullptr, nullptr, nullptr);
    gemm_bf16<EPI_NONE, 0><<<dim3((NCH_ * CVS_) / 128, B_ / 128, NR_), 256, 0, stream>>>(
        hnewb, NR_ * U_, U_, WcvT, (long)(NCH_ * CVS_) * U_, cvb, NR_ * NCH_ * CVS_, NCH_ * CVS_,
        nullptr, U_, nullptr, nullptr, nullptr, nullptr);
    // 7) communication attention -> ctx (bf16 perm)
    comm_attn<<<B_, 256, 0, stream>>>(ckb, cvb, cqb, maskf, ctxb);
    // 8) out = (ctx @ Wco[n] + h_new)*mask + hs*(1-mask)
    gemm_bf16<EPI_FINAL, 0><<<dim3(U_ / 128, B_ / 128, NR_), 256, 0, stream>>>(
        ctxb, NR_ * NCH_ * CVS_, NCH_ * CVS_, WcoT, (long)U_ * (NCH_ * CVS_),
        out, NR_ * U_, U_, nullptr, NCH_ * CVS_, nullptr, hnew, hs, maskf);

    (void)in_sizes; (void)n_in; (void)out_size; (void)ws_size;
}
